// Linformer_71433896067707
// MI455X (gfx1250) — matrix operand order repaired and checked by reading, not compiled, not run on hardware
//
#include <hip/hip_runtime.h>
#include <hip/hip_bf16.h>
#include <math.h>

// ---------------------------------------------------------------------------
// Linformer (B=2, N=2048, C=512, H=8, D=64, K=256, FF=2048, L=2) for gfx1250.
// Dense GEMMs -> bf16 WMMA (v_wmma_f32_16x16x32_bf16, f32 accum) with
//   double-buffered GLOBAL_LOAD_ASYNC_TO_LDS_B128 staging (ASYNCcnt pipeline).
// All GEMM B operands are pre-transposed to (N x K) bf16 at conversion time so
//   both A and B tiles stage as contiguous 16B async chunks (no LDS transpose).
// Channel-softmax attention -> wave32 shuffle-reduced VALU kernel.
// ---------------------------------------------------------------------------

typedef __bf16 bf16;
typedef __attribute__((ext_vector_type(16))) __bf16 v16bf;
typedef __attribute__((ext_vector_type(8)))  float  v8f;

#define Bc   2
#define Nc   2048
#define Cc   512
#define Hc   8
#define Kc   256
#define Dc   64
#define FFc  2048
#define Lc   2
#define Mrows (Bc * Nc)   // 4096 token rows

// ------------------------------ f32 -> bf16 --------------------------------
__global__ __launch_bounds__(256) void f32_to_bf16_kernel(
    const float* __restrict__ src, bf16* __restrict__ dst, int n)
{
    int i = blockIdx.x * 256 + threadIdx.x;
    if (i < n) dst[i] = (bf16)src[i];
}

// ---------------- f32 -> bf16 with per-matrix transpose --------------------
// nmat matrices of (Kd x Nd) row-major f32 -> (Nd x Kd) row-major bf16.
// dst is written linearly (coalesced); src reads are strided (weights: small,
// done once per launch).
__global__ __launch_bounds__(256) void f32_to_bf16_t_kernel(
    const float* __restrict__ src, bf16* __restrict__ dst,
    int Kd, int Nd, int nmat)
{
    int i = blockIdx.x * 256 + threadIdx.x;
    int per = Kd * Nd;
    if (i < per * nmat) {
        int m = i / per;
        int r = i - m * per;
        int n = r / Kd;
        int k = r - n * Kd;
        dst[i] = (bf16)src[(size_t)m * per + (size_t)k * Nd + n];
    }
}

// ------------------------------ WMMA GEMM ----------------------------------
// C[M,N] = A[M,K](bf16 row-major, lda=K) @ B[K,N] + bias, where B is supplied
// TRANSPOSED: Bt[N,K] bf16 row-major (ldbt=K).
// Block: 128 threads = 4 waves; block tile 64x64; wave tile 32x32; K-step 64.
// Staging: async global->LDS b128, double buffered, waited with asynccnt.
template <int GELU>
__global__ __launch_bounds__(128) void gemm_bf16_kernel(
    const bf16* __restrict__ A, const bf16* __restrict__ Bt,
    const float* __restrict__ bias, float* __restrict__ C,
    int M, int N, int K, int ldc)
{
    __shared__ bf16 As[2][64][72];   // [buf][m][k], 144B row stride (16B-mult)
    __shared__ bf16 Bs[2][64][72];   // [buf][n][k]

    const int tid  = threadIdx.x;
    const int lane = tid & 31;
    const int wave = tid >> 5;
    const int wm   = (wave >> 1) * 32;
    const int wn   = (wave & 1) * 32;
    const int tile_m = blockIdx.x * 64;
    const int tile_n = blockIdx.y * 64;

    // Issue one 64x64 bf16 tile of A and Bt as async 16B chunks (8 per thread).
    auto stage = [&](int k0, int buf) {
        #pragma unroll
        for (int c = tid; c < 512; c += 128) {
            const int row  = c >> 3;
            const int col8 = (c & 7) << 3;
            const unsigned long long ga =
                (unsigned long long)(size_t)&A[(size_t)(tile_m + row) * K + k0 + col8];
            const unsigned la = (unsigned)(size_t)&As[buf][row][col8];
            asm volatile("global_load_async_to_lds_b128 %0, %1, off"
                         :: "v"(la), "v"(ga) : "memory");
        }
        #pragma unroll
        for (int c = tid; c < 512; c += 128) {
            const int row  = c >> 3;
            const int col8 = (c & 7) << 3;
            const unsigned long long gb =
                (unsigned long long)(size_t)&Bt[(size_t)(tile_n + row) * K + k0 + col8];
            const unsigned lb = (unsigned)(size_t)&Bs[buf][row][col8];
            asm volatile("global_load_async_to_lds_b128 %0, %1, off"
                         :: "v"(lb), "v"(gb) : "memory");
        }
        // L2 prefetch of the tile after next.
        const int nk = k0 + 64;
        if (nk < K) {
            __builtin_prefetch(&A [(size_t)(tile_m + (tid >> 1)) * K + nk], 0, 1);
            __builtin_prefetch(&Bt[(size_t)(tile_n + (tid >> 1)) * K + nk], 0, 1);
        }
    };

    v8f acc[2][2];
    #pragma unroll
    for (int i = 0; i < 2; ++i)
        #pragma unroll
        for (int j = 0; j < 2; ++j)
            acc[i][j] = {};

    stage(0, 0);
    const int T = K >> 6;
    for (int t = 0; t < T; ++t) {
        __syncthreads();                       // prior compute done -> buffer free
        if (t + 1 < T) {
            stage((t + 1) << 6, (t + 1) & 1);
            // 8 new ops in flight; <=8 outstanding => tile t's 8 landed in LDS.
            asm volatile("s_wait_asynccnt 0x8" ::: "memory");
        } else {
            asm volatile("s_wait_asynccnt 0x0" ::: "memory");
        }
        __syncthreads();                       // tile t visible to all waves

        const int buf  = t & 1;
        const int arow = lane & 15;
        #pragma unroll
        for (int kc = 0; kc < 2; ++kc) {
            v16bf afrag[2], bfrag[2];
            // A frag (16x32): lanes 0-15 row=lane  K {0..7,16..23}(+kc*32);
            //                 lanes16-31 row=lane&15 K {8..15,24..31}(+kc*32).
            const int ac0 = kc * 32 + ((lane < 16) ? 0 : 8);
            #pragma unroll
            for (int mi = 0; mi < 2; ++mi) {
                const int r = wm + mi * 16 + arow;
                *((uint4*)&afrag[mi])     = *(const uint4*)&As[buf][r][ac0];
                *((uint4*)&afrag[mi] + 1) = *(const uint4*)&As[buf][r][ac0 + 16];
            }
            // B frag (32x16): lanes 0-15 col=lane K {0..7,16..23}(+kc*32);
            //                 lanes16-31 col=lane&15 K {8..15,24..31}(+kc*32).
            const int bk0 = kc * 32 + ((lane < 16) ? 0 : 8);
            #pragma unroll
            for (int ni = 0; ni < 2; ++ni) {
                const int nr = wn + ni * 16 + arow;
                *((uint4*)&bfrag[ni])     = *(const uint4*)&Bs[buf][nr][bk0];
                *((uint4*)&bfrag[ni] + 1) = *(const uint4*)&Bs[buf][nr][bk0 + 16];
            }
            #pragma unroll
            for (int mi = 0; mi < 2; ++mi)
                #pragma unroll
                for (int ni = 0; ni < 2; ++ni)
                    acc[mi][ni] = __builtin_amdgcn_wmma_f32_16x16x32_bf16(
                        false, afrag[mi], false, bfrag[ni],
                        (short)0, acc[mi][ni], false, false);
        }
    }

    // Epilogue: C/D layout: VGPR j -> M = j + 8*(lane/16), N = lane%16.
    const int rsub = (lane >> 4) << 3;
    const int csub = lane & 15;
    #pragma unroll
    for (int mi = 0; mi < 2; ++mi)
        #pragma unroll
        for (int ni = 0; ni < 2; ++ni)
            #pragma unroll
            for (int j = 0; j < 8; ++j) {
                int r = tile_m + wm + mi * 16 + rsub + j;
                int c = tile_n + wn + ni * 16 + csub;
                float v = acc[mi][ni][j];
                if (bias) v += bias[c];
                if (GELU) v = 0.5f * v * (1.0f + erff(v * 0.70710678118654752f));
                C[(size_t)r * ldc + c] = v;
            }
}

// --------------------- channel-softmax "attention" -------------------------
// One wave32 per (b,h,n); 2 channels per lane (D=64).
// out[b,n,h,:] = sum_k softmax_d(QW[b,n,h,:]*KW[b,k,h,:]*scale) * VW[b,k,h,:]
__global__ __launch_bounds__(256) void attn_kernel(
    const float* __restrict__ QW, const float* __restrict__ KW,
    const float* __restrict__ VW, float* __restrict__ out)
{
    const int g    = (blockIdx.x * 256 + threadIdx.x) >> 5;
    const int lane = threadIdx.x & 31;
    const int n  = g & (Nc - 1);
    const int bh = g >> 11;            // Nc = 2^11
    const int h  = bh & (Hc - 1);
    const int b  = bh >> 3;            // Hc = 2^3
    const float scale = 0.125f;        // 1/sqrt(64)

    const size_t qoff  = ((size_t)(b * Nc + n)) * Cc + h * Dc + 2 * lane;
    const size_t kbase = ((size_t)b * Nc) * Cc + h * Dc + 2 * lane;
    const float2 q = *(const float2*)&QW[qoff];

    float ax = 0.0f, ay = 0.0f;
    for (int k = 0; k < Kc; ++k) {
        const size_t off = kbase + (size_t)k * Cc;
        const float2 kw = *(const float2*)&KW[off];
        float s0 = q.x * kw.x * scale;
        float s1 = q.y * kw.y * scale;
        float m = fmaxf(s0, s1);
        #pragma unroll
        for (int o = 16; o; o >>= 1) m = fmaxf(m, __shfl_xor(m, o, 32));
        float e0 = __expf(s0 - m);
        float e1 = __expf(s1 - m);
        float s = e0 + e1;
        #pragma unroll
        for (int o = 16; o; o >>= 1) s += __shfl_xor(s, o, 32);
        float r = __builtin_amdgcn_rcpf(s);
        const float2 vw = *(const float2*)&VW[off];
        ax += e0 * r * vw.x;
        ay += e1 * r * vw.y;
    }
    float2 res; res.x = ax; res.y = ay;
    *(float2*)&out[qoff] = res;
}

// --------------------- residual add + LayerNorm ----------------------------
// One 256-thread block per token row (C=512 -> 2 elems/thread).
__global__ __launch_bounds__(256) void add_ln_kernel(
    const float* __restrict__ o, const float* __restrict__ res,
    const float* __restrict__ g, const float* __restrict__ be,
    float* __restrict__ dst)
{
    const int row = blockIdx.x;
    const int tid = threadIdx.x;
    const int c0  = 2 * tid;
    const size_t base = (size_t)row * Cc;

    const float2 a  = *(const float2*)&o[base + c0];
    const float2 rr = *(const float2*)&res[base + c0];
    float v0 = a.x + rr.x, v1 = a.y + rr.y;
    float s  = v0 + v1;
    float ss = v0 * v0 + v1 * v1;
    #pragma unroll
    for (int off = 16; off; off >>= 1) {
        s  += __shfl_xor(s, off, 32);
        ss += __shfl_xor(ss, off, 32);
    }
    __shared__ float sbuf[8], ssbuf[8];
    const int w = tid >> 5, lane = tid & 31;
    if (lane == 0) { sbuf[w] = s; ssbuf[w] = ss; }
    __syncthreads();
    if (w == 0) {
        float s2  = (lane < 8) ? sbuf[lane]  : 0.0f;
        float ss2 = (lane < 8) ? ssbuf[lane] : 0.0f;
        #pragma unroll
        for (int off = 4; off; off >>= 1) {
            s2  += __shfl_xor(s2, off, 32);
            ss2 += __shfl_xor(ss2, off, 32);
        }
        if (lane == 0) { sbuf[0] = s2; ssbuf[0] = ss2; }
    }
    __syncthreads();
    const float mean = sbuf[0] * (1.0f / Cc);
    const float var  = ssbuf[0] * (1.0f / Cc) - mean * mean;
    const float rstd = rsqrtf(var + 1e-5f);
    float2 y;
    y.x = (v0 - mean) * rstd * g[c0]     + be[c0];
    y.y = (v1 - mean) * rstd * g[c0 + 1] + be[c0 + 1];
    *(float2*)&dst[base + c0] = y;
}

// ------------------------------- host side ---------------------------------
extern "C" void kernel_launch(void* const* d_in, const int* in_sizes, int n_in,
                              void* d_out, int out_size, void* d_ws, size_t ws_size,
                              hipStream_t stream)
{
    (void)in_sizes; (void)n_in; (void)out_size; (void)ws_size;

    const float* x_in = (const float*)d_in[0];
    const float* to_q = (const float*)d_in[1];
    const float* to_k = (const float*)d_in[2];
    const float* to_v = (const float*)d_in[3];
    const float* wq   = (const float*)d_in[4];
    const float* bq   = (const float*)d_in[5];
    const float* wk   = (const float*)d_in[6];
    const float* bk   = (const float*)d_in[7];
    const float* wv   = (const float*)d_in[8];
    const float* bv   = (const float*)d_in[9];
    const float* wo   = (const float*)d_in[10];
    const float* bo   = (const float*)d_in[11];
    const float* g1   = (const float*)d_in[12];
    const float* be1  = (const float*)d_in[13];
    const float* w1   = (const float*)d_in[14];
    const float* bb1  = (const float*)d_in[15];
    const float* w2   = (const float*)d_in[16];
    const float* bb2  = (const float*)d_in[17];
    const float* g2   = (const float*)d_in[18];
    const float* be2  = (const float*)d_in[19];

    char* p = (char*)d_ws;
    auto alloc = [&](size_t bytes) -> void* {
        void* r = (void*)p;
        p += (bytes + 255) & ~(size_t)255;
        return r;
    };

    // bf16 weights: A-side plain, B-side transposed (N x K).
    bf16* toq_bf = (bf16*)alloc(sizeof(bf16) * Lc * Cc * Dc);       // A operand
    bf16* tok_bf = (bf16*)alloc(sizeof(bf16) * Lc * Cc * Dc);
    bf16* tov_bf = (bf16*)alloc(sizeof(bf16) * Lc * Cc * Dc);
    bf16* wqT_bf = (bf16*)alloc(sizeof(bf16) * Lc * Hc * Dc * Dc);  // B^T
    bf16* wkT_bf = (bf16*)alloc(sizeof(bf16) * Lc * Hc * Dc * Dc);
    bf16* wvT_bf = (bf16*)alloc(sizeof(bf16) * Lc * Hc * Dc * Dc);
    bf16* woT_bf = (bf16*)alloc(sizeof(bf16) * Lc * Cc * Cc);
    bf16* w1T_bf = (bf16*)alloc(sizeof(bf16) * Lc * Cc * FFc);
    bf16* w2T_bf = (bf16*)alloc(sizeof(bf16) * Lc * FFc * Cc);
    // fused per-layer head projections (C x HD) f32, then transposed bf16
    float* Wqcat = (float*)alloc(sizeof(float) * Cc * Cc);
    float* Wkcat = (float*)alloc(sizeof(float) * Cc * Cc);
    float* Wvcat = (float*)alloc(sizeof(float) * Cc * Cc);
    bf16* WqcatT_bf = (bf16*)alloc(sizeof(bf16) * Cc * Cc);
    bf16* WkcatT_bf = (bf16*)alloc(sizeof(bf16) * Cc * Cc);
    bf16* WvcatT_bf = (bf16*)alloc(sizeof(bf16) * Cc * Cc);
    // activations
    float* xbuf   = (float*)alloc(sizeof(float) * Mrows * Cc);
    bf16*  act_bf = (bf16*)alloc(sizeof(bf16) * Mrows * FFc);   // biggest bf16 act
    float* QW     = (float*)alloc(sizeof(float) * Mrows * Cc);
    float* KW     = (float*)alloc(sizeof(float) * Mrows * Cc);
    float* VW     = (float*)alloc(sizeof(float) * Mrows * Cc);
    float* attn   = (float*)alloc(sizeof(float) * Mrows * Cc);
    float* h1     = (float*)alloc(sizeof(float) * Mrows * FFc);
    float* obuf   = (float*)alloc(sizeof(float) * Mrows * Cc);

    auto conv = [&](const float* s, bf16* d, int n) {
        f32_to_bf16_kernel<<<(n + 255) / 256, 256, 0, stream>>>(s, d, n);
    };
    auto convT = [&](const float* s, bf16* d, int Kd, int Nd, int nmat) {
        int n = Kd * Nd * nmat;
        f32_to_bf16_t_kernel<<<(n + 255) / 256, 256, 0, stream>>>(s, d, Kd, Nd, nmat);
    };
    auto gemm = [&](const bf16* A, const bf16* Bt, const float* bias, float* Cm,
                    int M, int N, int K, int ldc, bool gelu) {
        dim3 grid(M / 64, N / 64), block(128);
        if (gelu)
            gemm_bf16_kernel<1><<<grid, block, 0, stream>>>(A, Bt, bias, Cm, M, N, K, ldc);
        else
            gemm_bf16_kernel<0><<<grid, block, 0, stream>>>(A, Bt, bias, Cm, M, N, K, ldc);
    };

    // x <- input
    hipMemcpyAsync(xbuf, x_in, sizeof(float) * Mrows * Cc,
                   hipMemcpyDeviceToDevice, stream);

    // Weights -> bf16 once. A-side plain; B-side transposed per matrix.
    conv(to_q, toq_bf, Lc * Cc * Dc);
    conv(to_k, tok_bf, Lc * Cc * Dc);
    conv(to_v, tov_bf, Lc * Cc * Dc);
    convT(wq, wqT_bf, Dc, Dc, Lc * Hc);
    convT(wk, wkT_bf, Dc, Dc, Lc * Hc);
    convT(wv, wvT_bf, Dc, Dc, Lc * Hc);
    convT(wo, woT_bf, Cc, Cc, Lc);
    convT(w1, w1T_bf, Cc, FFc, Lc);
    convT(w2, w2T_bf, FFc, Cc, Lc);

    for (int l = 0; l < Lc; ++l) {
        // Fuse per-head weights: Wcat[:, h*64+d] = to_*[l] @ w*[l,h]
        for (int h = 0; h < Hc; ++h) {
            gemm(toq_bf + l * Cc * Dc, wqT_bf + (l * Hc + h) * Dc * Dc, nullptr,
                 Wqcat + h * Dc, Cc, Dc, Dc, Cc, false);
            gemm(tok_bf + l * Cc * Dc, wkT_bf + (l * Hc + h) * Dc * Dc, nullptr,
                 Wkcat + h * Dc, Cc, Dc, Dc, Cc, false);
            gemm(tov_bf + l * Cc * Dc, wvT_bf + (l * Hc + h) * Dc * Dc, nullptr,
                 Wvcat + h * Dc, Cc, Dc, Dc, Cc, false);
        }
        convT(Wqcat, WqcatT_bf, Cc, Cc, 1);
        convT(Wkcat, WkcatT_bf, Cc, Cc, 1);
        convT(Wvcat, WvcatT_bf, Cc, Cc, 1);

        // QW/KW/VW = x @ Wcat + bias   (bias bq[l] is already (H,D)-concat)
        conv(xbuf, act_bf, Mrows * Cc);
        gemm(act_bf, WqcatT_bf, bq + l * Hc * Dc, QW, Mrows, Cc, Cc, Cc, false);
        gemm(act_bf, WkcatT_bf, bk + l * Hc * Dc, KW, Mrows, Cc, Cc, Cc, false);
        gemm(act_bf, WvcatT_bf, bv + l * Hc * Dc, VW, Mrows, Cc, Cc, Cc, false);

        // channel-softmax attention (E=eye(N,K) -> KW/VW row-slice via index)
        attn_kernel<<<(Bc * Hc * Nc * 32) / 256, 256, 0, stream>>>(QW, KW, VW, attn);

        // output projection + residual LN
        conv(attn, act_bf, Mrows * Cc);
        gemm(act_bf, woT_bf + l * Cc * Cc, bo + l * Cc, obuf, Mrows, Cc, Cc, Cc, false);
        add_ln_kernel<<<Mrows, 256, 0, stream>>>(obuf, xbuf, g1 + l * Cc, be1 + l * Cc, xbuf);

        // FFN: gelu(x@w1+b1)@w2+b2, residual LN
        conv(xbuf, act_bf, Mrows * Cc);
        gemm(act_bf, w1T_bf + l * Cc * FFc, bb1 + l * FFc, h1, Mrows, FFc, Cc, FFc, true);
        conv(h1, act_bf, Mrows * FFc);
        gemm(act_bf, w2T_bf + l * FFc * Cc, bb2 + l * Cc, obuf, Mrows, Cc, FFc, Cc, false);
        float* dst = (l == Lc - 1) ? (float*)d_out : xbuf;
        add_ln_kernel<<<Mrows, 256, 0, stream>>>(obuf, xbuf, g2 + l * Cc, be2 + l * Cc, dst);
    }
}
